// GAT_xyz_420906795146
// MI455X (gfx1250) — matrix operand order repaired and checked by reading, not compiled, not run on hardware
//
#include <hip/hip_runtime.h>
#include <stdint.h>

// Problem constants (from the reference)
#define NN 50000          // nodes
#define EE 400000         // edges (before self loops)
#define ET (EE + NN)      // total edges incl. self loops
#define FF 128            // input features
#define HH 8              // heads
#define C1 64             // conv1 per-head channels
#define C2 32             // conv2 per-head channels (= num classes)
#define NEG_SLOPE 0.2f

typedef __attribute__((ext_vector_type(16))) __bf16 v16bf;
typedef __attribute__((ext_vector_type(8)))  float  v8f;

// ---------- helpers ----------
__device__ __forceinline__ unsigned short f2bf(float f) {
  // round-to-nearest-even fp32 -> bf16
  unsigned u = __float_as_uint(f);
  u += 0x7fffu + ((u >> 16) & 1u);
  return (unsigned short)(u >> 16);
}
// monotone order-preserving float <-> uint encoding (for atomic max on floats)
__device__ __forceinline__ unsigned encOrd(float f) {
  unsigned u = __float_as_uint(f);
  return (u & 0x80000000u) ? ~u : (u | 0x80000000u);
}
__device__ __forceinline__ float decOrd(unsigned e) {
  return __uint_as_float((e & 0x80000000u) ? (e & 0x7fffffffu) : ~e);
}
// Hardware fp32 atomic add (GLOBAL_ATOMIC_ADD_F32, no return) — avoids CAS loop.
__device__ __forceinline__ void atomAddF(float* p, float v) {
  asm volatile("global_atomic_add_f32 %0, %1, off" :: "v"(p), "v"(v) : "memory");
}

// ---------- fp32 -> bf16 convert ----------
__global__ void cvt_bf16_kernel(const float* __restrict__ in,
                                unsigned short* __restrict__ out, int n) {
  int i = blockIdx.x * blockDim.x + threadIdx.x;
  if (i < n) out[i] = f2bf(in[i]);
}

// ---------- pack a row-major fp32 weight [K x Nw] into WMMA B-fragment order ----
// B fragment (this code's contiguous k order, not the ISA table order): lane = n + 16*(k>=16), element t = k&15.
// Packed layout: [jt][kt][lane][t], jt = 16-col tile, kt = 32-K tile.
__global__ void pack_b_kernel(const float* __restrict__ W,
                              unsigned short* __restrict__ Bp, int K, int Nw) {
  int i = blockIdx.x * blockDim.x + threadIdx.x;
  if (i >= K * Nw) return;
  int k = i / Nw, n = i % Nw;
  int kt = k >> 5, kk = k & 31;
  int jt = n >> 4, nn = n & 15;
  int lane = nn + ((kk >> 4) << 4);
  int t = kk & 15;
  int ktiles = K >> 5;
  size_t off = ((((size_t)jt * ktiles + kt) * 32 + lane) << 4) + t;
  Bp[off] = f2bf(W[i]);
}

// ---------- bf16 WMMA GEMM: C[M x Nw] = A[M x K] * B (+bias) ----------
// One wave -> 16 rows x 32 cols (two 16x16 WMMA tiles, A fragment reused).
// M multiple of 16 (50000 = 3125*16), K multiple of 32, Nw multiple of 32.
__global__ void __launch_bounds__(128)
gemm_bf16_wmma_kernel(const unsigned short* __restrict__ A,
                      const unsigned short* __restrict__ Bp,
                      float* __restrict__ C, const float* __restrict__ bias,
                      int M, int K, int Nw) {
  const int lane = threadIdx.x & 31;
  const int ti = blockIdx.x * 4 + (threadIdx.x >> 5);  // 16-row tile
  const int tj = blockIdx.y;                           // 32-col slab
  const int mTiles = M >> 4;
  if (ti >= mTiles) return;
  const int g = lane >> 4;     // half-wave group
  const int m = lane & 15;     // row within tile (A) / col within tile (B,C)
  const int ktiles = K >> 5;

  union Frag { uint4 u[2]; v16bf v; };
  v8f c0 = {}; v8f c1 = {};

  // A (this code's contiguous k order, matching packed B): lane g*16+m holds K = 32*kt + 16g + [0..15].
  const unsigned short* arow = A + (size_t)((ti << 4) + m) * K + 16 * g;
  const size_t jstride = (size_t)ktiles * 32 * 16;   // elements per 16-col tile
  const unsigned short* b0p = Bp + (size_t)(2 * tj) * jstride + ((size_t)lane << 4);
  const unsigned short* b1p = b0p + jstride;

  for (int kt = 0; kt < ktiles; ++kt) {
    Frag a, b0, b1;
    a.u[0]  = *(const uint4*)(arow + kt * 32);
    a.u[1]  = *(const uint4*)(arow + kt * 32 + 8);
    b0.u[0] = *(const uint4*)(b0p + (size_t)kt * 512);
    b0.u[1] = *(const uint4*)(b0p + (size_t)kt * 512 + 8);
    b1.u[0] = *(const uint4*)(b1p + (size_t)kt * 512);
    b1.u[1] = *(const uint4*)(b1p + (size_t)kt * 512 + 8);
    c0 = __builtin_amdgcn_wmma_f32_16x16x32_bf16(false, a.v, false, b0.v,
                                                 (short)0, c0, false, false);
    c1 = __builtin_amdgcn_wmma_f32_16x16x32_bf16(false, a.v, false, b1.v,
                                                 (short)0, c1, false, false);
  }
  // C/D layout: lane 16g+n holds D[r+8g][n] in element r.
  const int col = (tj << 5) + m;
  const float bv0 = bias ? bias[col]      : 0.0f;
  const float bv1 = bias ? bias[col + 16] : 0.0f;
  for (int r = 0; r < 8; ++r) {
    float* crow = C + (size_t)((ti << 4) + 8 * g + r) * Nw + col;
    crow[0]  = c0[r] + bv0;
    crow[16] = c1[r] + bv1;
  }
}

// ---------- edge pass 1: scores + segment max ----------
template <int CC>
__global__ void edge_score_kernel(const float* __restrict__ xl,
                                  const float* __restrict__ xr,
                                  const int* __restrict__ ei,
                                  const float* __restrict__ att,   // [H, CC]
                                  float* __restrict__ eBuf,        // [ET*H]
                                  unsigned* __restrict__ mEnc) {   // [N*H]
  int t = blockIdx.x * blockDim.x + threadIdx.x;
  if (t >= ET * HH) return;
  int e = t >> 3, h = t & 7;
  int s, d;
  if (e < EE) { s = ei[e]; d = ei[EE + e]; } else { s = d = e - EE; }
  const int ld = HH * CC;
  const float4* pl = (const float4*)(xl + (size_t)s * ld + h * CC);
  const float4* pr = (const float4*)(xr + (size_t)d * ld + h * CC);
  const float4* pa = (const float4*)(att + h * CC);
  float acc = 0.0f;
#pragma unroll
  for (int c4 = 0; c4 < (CC >> 2); ++c4) {
    float4 a = pl[c4], b = pr[c4], w = pa[c4];
    float v;
    v = a.x + b.x; acc += w.x * (v > 0.0f ? v : NEG_SLOPE * v);
    v = a.y + b.y; acc += w.y * (v > 0.0f ? v : NEG_SLOPE * v);
    v = a.z + b.z; acc += w.z * (v > 0.0f ? v : NEG_SLOPE * v);
    v = a.w + b.w; acc += w.w * (v > 0.0f ? v : NEG_SLOPE * v);
  }
  eBuf[t] = acc;
  atomicMax(&mEnc[d * HH + h], encOrd(acc));
}

// ---------- edge pass 2: exp, denom, weighted scatter ----------
template <int CC>
__global__ void edge_aggregate_kernel(const float* __restrict__ xl,
                                      const int* __restrict__ ei,
                                      const float* __restrict__ eBuf,
                                      const unsigned* __restrict__ mEnc,
                                      float* __restrict__ denom,    // [N*H]
                                      float* __restrict__ acc) {    // [N*H*CC]
  int t = blockIdx.x * blockDim.x + threadIdx.x;
  if (t >= ET * HH) return;
  int e = t >> 3, h = t & 7;
  int s, d;
  if (e < EE) { s = ei[e]; d = ei[EE + e]; } else { s = d = e - EE; }
  float ex = __expf(eBuf[t] - decOrd(mEnc[d * HH + h]));
  atomAddF(&denom[d * HH + h], ex);
  const int ld = HH * CC;
  const float* pl = xl + (size_t)s * ld + h * CC;
  float* po = acc + (size_t)d * ld + h * CC;
#pragma unroll
  for (int c = 0; c < CC; c += 4) {
    float4 v = *(const float4*)(pl + c);
    atomAddF(po + c + 0, ex * v.x);
    atomAddF(po + c + 1, ex * v.y);
    atomAddF(po + c + 2, ex * v.z);
    atomAddF(po + c + 3, ex * v.w);
  }
}

// ---------- layer1 epilogue: head-mean/softmax-normalize + bias + relu ----------
__global__ void node_finish1_kernel(const float* __restrict__ acc,
                                    const float* __restrict__ denom,
                                    const float* __restrict__ bias,
                                    float* __restrict__ hOut,
                                    unsigned short* __restrict__ hBf) {
  int i = blockIdx.x * blockDim.x + threadIdx.x;
  if (i >= NN * C1) return;
  int n = i >> 6, c = i & 63;
  float s = 0.0f;
#pragma unroll
  for (int h = 0; h < HH; ++h)
    s += acc[(size_t)n * (HH * C1) + h * C1 + c] / denom[n * HH + h];
  float v = s * (1.0f / HH) + bias[c];
  v = v > 0.0f ? v : 0.0f;
  hOut[i] = v;
  hBf[i] = f2bf(v);
}

// ---------- layer2 epilogue: head-mean + bias + residual ----------
__global__ void node_finish2_kernel(const float* __restrict__ acc,
                                    const float* __restrict__ denom,
                                    const float* __restrict__ bias,
                                    const float* __restrict__ resid,
                                    float* __restrict__ out) {
  int i = blockIdx.x * blockDim.x + threadIdx.x;
  if (i >= NN * C2) return;
  int n = i >> 5, c = i & 31;
  float s = 0.0f;
#pragma unroll
  for (int h = 0; h < HH; ++h)
    s += acc[(size_t)n * (HH * C2) + h * C2 + c] / denom[n * HH + h];
  out[i] = s * (1.0f / HH) + bias[c] + resid[i];
}

// =======================================================================
extern "C" void kernel_launch(void* const* d_in, const int* in_sizes, int n_in,
                              void* d_out, int out_size, void* d_ws, size_t ws_size,
                              hipStream_t stream) {
  (void)in_sizes; (void)n_in; (void)out_size; (void)ws_size;
  const float* x      = (const float*)d_in[0];
  const int*   ei     = (const int*)  d_in[1];   // [2,E]: row0=src, row1=dst
  /* xyz = d_in[2] is unused by the reference */
  const float* lin1_w = (const float*)d_in[3];
  const float* lin1_b = (const float*)d_in[4];
  const float* wl1    = (const float*)d_in[5];
  const float* wr1    = (const float*)d_in[6];
  const float* att1   = (const float*)d_in[7];
  const float* b1     = (const float*)d_in[8];
  const float* wl2    = (const float*)d_in[9];
  const float* wr2    = (const float*)d_in[10];
  const float* att2   = (const float*)d_in[11];
  const float* b2     = (const float*)d_in[12];
  float* out = (float*)d_out;

  // ---- workspace layout (bump allocator, 256B aligned) ----
  char* ws = (char*)d_ws;
  size_t off = 0;
  auto alloc = [&](size_t bytes) -> char* {
    char* p = ws + off;
    off += (bytes + 255) & ~(size_t)255;
    return p;
  };
  unsigned short* xb   = (unsigned short*)alloc((size_t)NN * FF * 2);
  unsigned short* pWL1 = (unsigned short*)alloc((size_t)FF * (HH * C1) * 2);
  unsigned short* pWR1 = (unsigned short*)alloc((size_t)FF * (HH * C1) * 2);
  unsigned short* pLIN = (unsigned short*)alloc((size_t)FF * C2 * 2);
  unsigned short* pWL2 = (unsigned short*)alloc((size_t)C1 * (HH * C2) * 2);
  unsigned short* pWR2 = (unsigned short*)alloc((size_t)C1 * (HH * C2) * 2);
  float* xl   = (float*)alloc((size_t)NN * HH * C1 * 4);  // layer2 xl aliases (smaller)
  float* xr   = (float*)alloc((size_t)NN * HH * C1 * 4);
  float* accb = (float*)alloc((size_t)NN * HH * C1 * 4);
  float* eBuf = (float*)alloc((size_t)ET * HH * 4);
  unsigned* mEnc = (unsigned*)alloc((size_t)NN * HH * 4);
  float* denom   = (float*)   alloc((size_t)NN * HH * 4);
  float* resid   = (float*)   alloc((size_t)NN * C2 * 4);
  float* hbuf    = (float*)   alloc((size_t)NN * C1 * 4);
  unsigned short* hbf = (unsigned short*)alloc((size_t)NN * C1 * 2);

  const int mTiles = NN / 16;               // 3125 exactly
  const dim3 gw((mTiles + 3) / 4, 1, 1);
  const int edgeThreads = ET * HH;
  const int edgeBlocks = (edgeThreads + 255) / 256;

  // ---- precision prep: activations & weights to bf16 (WMMA A/B) ----
  cvt_bf16_kernel<<<(NN * FF + 255) / 256, 256, 0, stream>>>(x, xb, NN * FF);
  pack_b_kernel<<<(FF * HH * C1 + 255) / 256, 256, 0, stream>>>(wl1, pWL1, FF, HH * C1);
  pack_b_kernel<<<(FF * HH * C1 + 255) / 256, 256, 0, stream>>>(wr1, pWR1, FF, HH * C1);
  pack_b_kernel<<<(FF * C2 + 255) / 256, 256, 0, stream>>>(lin1_w, pLIN, FF, C2);
  pack_b_kernel<<<(C1 * HH * C2 + 255) / 256, 256, 0, stream>>>(wl2, pWL2, C1, HH * C2);
  pack_b_kernel<<<(C1 * HH * C2 + 255) / 256, 256, 0, stream>>>(wr2, pWR2, C1, HH * C2);

  // ---- layer 1 GEMMs (WMMA): xl1 = x@wl1, xr1 = x@wr1, resid = x@lin1_w + b ----
  gemm_bf16_wmma_kernel<<<dim3(gw.x, (HH * C1) / 32), 128, 0, stream>>>(
      xb, pWL1, xl, nullptr, NN, FF, HH * C1);
  gemm_bf16_wmma_kernel<<<dim3(gw.x, (HH * C1) / 32), 128, 0, stream>>>(
      xb, pWR1, xr, nullptr, NN, FF, HH * C1);
  gemm_bf16_wmma_kernel<<<dim3(gw.x, C2 / 32), 128, 0, stream>>>(
      xb, pLIN, resid, lin1_b, NN, FF, C2);

  // ---- layer 1 edge softmax + aggregation ----
  hipMemsetAsync(mEnc, 0, (size_t)NN * HH * 4, stream);   // 0 < encOrd(-inf): floor below any score
  hipMemsetAsync(denom, 0, (size_t)NN * HH * 4, stream);
  hipMemsetAsync(accb, 0, (size_t)NN * HH * C1 * 4, stream);
  edge_score_kernel<C1><<<edgeBlocks, 256, 0, stream>>>(xl, xr, ei, att1, eBuf, mEnc);
  edge_aggregate_kernel<C1><<<edgeBlocks, 256, 0, stream>>>(xl, ei, eBuf, mEnc, denom, accb);
  node_finish1_kernel<<<(NN * C1 + 255) / 256, 256, 0, stream>>>(accb, denom, b1, hbuf, hbf);

  // ---- layer 2 GEMMs (reuse xl/xr buffers; K=64, Nw=256) ----
  gemm_bf16_wmma_kernel<<<dim3(gw.x, (HH * C2) / 32), 128, 0, stream>>>(
      hbf, pWL2, xl, nullptr, NN, C1, HH * C2);
  gemm_bf16_wmma_kernel<<<dim3(gw.x, (HH * C2) / 32), 128, 0, stream>>>(
      hbf, pWR2, xr, nullptr, NN, C1, HH * C2);

  // ---- layer 2 edge softmax + aggregation ----
  hipMemsetAsync(mEnc, 0, (size_t)NN * HH * 4, stream);
  hipMemsetAsync(denom, 0, (size_t)NN * HH * 4, stream);
  hipMemsetAsync(accb, 0, (size_t)NN * HH * C2 * 4, stream);
  edge_score_kernel<C2><<<edgeBlocks, 256, 0, stream>>>(xl, xr, ei, att2, eBuf, mEnc);
  edge_aggregate_kernel<C2><<<edgeBlocks, 256, 0, stream>>>(xl, ei, eBuf, mEnc, denom, accb);
  node_finish2_kernel<<<(NN * C2 + 255) / 256, 256, 0, stream>>>(accb, denom, b2, resid, out);
}
